// RosaBase_63299228008847
// MI455X (gfx1250) — matrix operand order repaired and checked by reading, not compiled, not run on hardware
//
#include <hip/hip_runtime.h>
#include <hip/hip_bf16.h>

// ---------------------------------------------------------------------------
// ROSA forward for MI455X (gfx1250): 4 bf16 WMMA GEMMs + VALU window/softmax.
// Shapes fixed by the reference: B=2, S=4096, D=1024, H=128, BQ=BV=8, W=8.
// GEMM: one wave computes a 32x64 tile (2 A-frags x 4 B-frags = 8 WMMAs/step)
// to raise arithmetic intensity to ~21 FLOP/byte from cache.
// ---------------------------------------------------------------------------

typedef __attribute__((ext_vector_type(16))) __bf16 v16bf;
typedef __attribute__((ext_vector_type(8)))  __bf16 v8bf;
typedef __attribute__((ext_vector_type(8)))  float  v8f;

#define ROWS   8192   // B*S
#define DMODEL 1024   // D == HQ == HV
#define NHEADS 128
#define NBITS  8
#define SEQ    4096
#define WIN    8

// ---------------------------- fp32 -> bf16 ---------------------------------
__global__ __launch_bounds__(256) void cvt_f32_bf16(const float* __restrict__ in,
                                                    __bf16* __restrict__ out, int n) {
  int i = (blockIdx.x * 256 + threadIdx.x) * 4;
  if (i + 3 < n) {
    float4 v = *(const float4*)(in + i);
    out[i + 0] = (__bf16)v.x;
    out[i + 1] = (__bf16)v.y;
    out[i + 2] = (__bf16)v.z;
    out[i + 3] = (__bf16)v.w;
  } else {
    for (; i < n; ++i) out[i] = (__bf16)in[i];
  }
}

// ------------------------- bf16 WMMA GEMM ----------------------------------
// C[M,N] = act( A[M,K] @ Bw[N,K]^T ), act = sigmoid if do_sigmoid.
__device__ __forceinline__ v16bf make_frag(v8bf lo, v8bf hi) {
  v16bf f;
#pragma unroll
  for (int i = 0; i < 8; ++i) { f[i] = lo[i]; f[i + 8] = hi[i]; }
  return f;
}

__global__ __launch_bounds__(256) void gemm_wmma_bf16(
    const __bf16* __restrict__ A, const __bf16* __restrict__ Bw,
    float* __restrict__ C, int M, int N, int K, int do_sigmoid) {
  const int lane = threadIdx.x & 31;
  const int wid  = blockIdx.x * 8 + (threadIdx.x >> 5);
  const int ntile = N >> 6;                 // 64-column tiles
  const int tm = wid / ntile;               // 32-row tiles
  const int tn = wid % ntile;
  if (tm * 32 >= M) return;                 // wave-uniform exit (EXEC stays all-1s)

  const int ml = lane & 15;                 // A row / B column within 16-tile
  const int h  = lane >> 4;                 // lane half selects K sub-block
  const size_t row0 = (size_t)tm * 32;
  const size_t col0 = (size_t)tn * 64;

  // A 16x32 bf16 fragment: lane half h holds K = h*16+0..15 (two b128 loads)
  const __bf16* Ap0 = A + (row0 +  0 + ml) * (size_t)K + h * 16;
  const __bf16* Ap1 = A + (row0 + 16 + ml) * (size_t)K + h * 16;
  // B 32x16 bf16 fragment: lane half h holds K = h*16+0..15 (two b128 loads)
  const __bf16* Bp0 = Bw + (col0 +  0 + ml) * (size_t)K + h * 16;
  const __bf16* Bp1 = Bw + (col0 + 16 + ml) * (size_t)K + h * 16;
  const __bf16* Bp2 = Bw + (col0 + 32 + ml) * (size_t)K + h * 16;
  const __bf16* Bp3 = Bw + (col0 + 48 + ml) * (size_t)K + h * 16;

  v8f acc[2][4];
#pragma unroll
  for (int r = 0; r < 2; ++r)
#pragma unroll
    for (int c = 0; c < 4; ++c) acc[r][c] = (v8f){};

  for (int k = 0; k < K; k += 32) {
    __builtin_prefetch(Ap0 + k + 256, 0, 1);   // global_prefetch_b8
    __builtin_prefetch(Ap1 + k + 256, 0, 1);
    v16bf a0 = make_frag(*(const v8bf*)(Ap0 + k), *(const v8bf*)(Ap0 + k + 8));
    v16bf a1 = make_frag(*(const v8bf*)(Ap1 + k), *(const v8bf*)(Ap1 + k + 8));
    v16bf b0 = make_frag(*(const v8bf*)(Bp0 + k), *(const v8bf*)(Bp0 + k + 8));
    v16bf b1 = make_frag(*(const v8bf*)(Bp1 + k), *(const v8bf*)(Bp1 + k + 8));
    v16bf b2 = make_frag(*(const v8bf*)(Bp2 + k), *(const v8bf*)(Bp2 + k + 8));
    v16bf b3 = make_frag(*(const v8bf*)(Bp3 + k), *(const v8bf*)(Bp3 + k + 8));
    acc[0][0] = __builtin_amdgcn_wmma_f32_16x16x32_bf16(false, a0, false, b0, (short)0, acc[0][0], false, false);
    acc[0][1] = __builtin_amdgcn_wmma_f32_16x16x32_bf16(false, a0, false, b1, (short)0, acc[0][1], false, false);
    acc[0][2] = __builtin_amdgcn_wmma_f32_16x16x32_bf16(false, a0, false, b2, (short)0, acc[0][2], false, false);
    acc[0][3] = __builtin_amdgcn_wmma_f32_16x16x32_bf16(false, a0, false, b3, (short)0, acc[0][3], false, false);
    acc[1][0] = __builtin_amdgcn_wmma_f32_16x16x32_bf16(false, a1, false, b0, (short)0, acc[1][0], false, false);
    acc[1][1] = __builtin_amdgcn_wmma_f32_16x16x32_bf16(false, a1, false, b1, (short)0, acc[1][1], false, false);
    acc[1][2] = __builtin_amdgcn_wmma_f32_16x16x32_bf16(false, a1, false, b2, (short)0, acc[1][2], false, false);
    acc[1][3] = __builtin_amdgcn_wmma_f32_16x16x32_bf16(false, a1, false, b3, (short)0, acc[1][3], false, false);
  }

  // C/D layout: VGPR r -> row = rowbase + h*8 + r, col = col0 + (lane&15)
#pragma unroll
  for (int g = 0; g < 2; ++g) {
    float* Cp = C + (row0 + (size_t)g * 16 + (size_t)h * 8) * (size_t)N + col0 + ml;
#pragma unroll
    for (int r = 0; r < 8; ++r) {
      float x0 = acc[g][0][r], x1 = acc[g][1][r], x2 = acc[g][2][r], x3 = acc[g][3][r];
      if (do_sigmoid) {
        x0 = 1.f / (1.f + __expf(-x0));
        x1 = 1.f / (1.f + __expf(-x1));
        x2 = 1.f / (1.f + __expf(-x2));
        x3 = 1.f / (1.f + __expf(-x3));
      }
      Cp[(size_t)r * N +  0] = x0;
      Cp[(size_t)r * N + 16] = x1;
      Cp[(size_t)r * N + 32] = x2;
      Cp[(size_t)r * N + 48] = x3;
    }
  }
}

// ------------------ window scores + softmax + gate -------------------------
// One thread per (b*S+s, head). Q/K/V are post-sigmoid fp32 [ROWS, 1024].
// Writes gated activations as bf16 (input of the final GEMM).
__global__ __launch_bounds__(256) void rosa_window_gate(
    const float* __restrict__ Q, const float* __restrict__ Kb,
    const float* __restrict__ Vb,
    const float* __restrict__ emb0, const float* __restrict__ emb1,
    __bf16* __restrict__ G) {
  const int idx = blockIdx.x * 256 + threadIdx.x;   // [0, ROWS*NHEADS)
  const int h   = idx & (NHEADS - 1);
  const int row = idx >> 7;
  const int s   = row & (SEQ - 1);
  const size_t base = (size_t)row * DMODEL + (size_t)h * NBITS;

  float4 q0 = *(const float4*)(Q + base);
  float4 q1 = *(const float4*)(Q + base + 4);
  float q[NBITS] = {q0.x, q0.y, q0.z, q0.w, q1.x, q1.y, q1.z, q1.w};
  float qsum = q[0] + q[1] + q[2] + q[3] + q[4] + q[5] + q[6] + q[7];

  const float scale = 0.3535533905932738f;  // 1/sqrt(8)
  float sc[WIN];
#pragma unroll
  for (int w = 0; w < WIN; ++w) {
    if (w <= s) {
      const float* kp = Kb + base - (size_t)w * DMODEL;
      float4 k0 = *(const float4*)kp;
      float4 k1 = *(const float4*)(kp + 4);
      float dot = q[0]*k0.x + q[1]*k0.y + q[2]*k0.z + q[3]*k0.w
                + q[4]*k1.x + q[5]*k1.y + q[6]*k1.z + q[7]*k1.w;
      float ksum = k0.x + k0.y + k0.z + k0.w + k1.x + k1.y + k1.z + k1.w;
      sc[w] = (8.f + 2.f * dot - qsum - ksum) * scale;
    } else {
      sc[w] = -1e30f;
    }
  }
  float mx = sc[0];
#pragma unroll
  for (int w = 1; w < WIN; ++w) mx = fmaxf(mx, sc[w]);
  float e[WIN], ssum = 0.f;
#pragma unroll
  for (int w = 0; w < WIN; ++w) { e[w] = __expf(sc[w] - mx); ssum += e[w]; }
  const float inv = 1.f / ssum;

  float out[NBITS] = {0.f, 0.f, 0.f, 0.f, 0.f, 0.f, 0.f, 0.f};
#pragma unroll
  for (int w = 0; w < WIN; ++w) {
    if (w <= s) {
      const float wg = e[w] * inv;
      const float* vp = Vb + base - (size_t)w * DMODEL;
      float4 v0 = *(const float4*)vp;
      float4 v1 = *(const float4*)(vp + 4);
      out[0] += wg * v0.x; out[1] += wg * v0.y; out[2] += wg * v0.z; out[3] += wg * v0.w;
      out[4] += wg * v1.x; out[5] += wg * v1.y; out[6] += wg * v1.z; out[7] += wg * v1.w;
    }
  }
#pragma unroll
  for (int c = 0; c < NBITS; ++c) {
    const int j = h * NBITS + c;
    const float g = emb1[j] * out[c] + emb0[j] * (1.f - out[c]);
    G[base + c] = (__bf16)g;
  }
}

// ---------------------------------------------------------------------------
extern "C" void kernel_launch(void* const* d_in, const int* in_sizes, int n_in,
                              void* d_out, int out_size, void* d_ws, size_t ws_size,
                              hipStream_t stream) {
  const float* X    = (const float*)d_in[0];   // [2,4096,1024]
  const float* Wq   = (const float*)d_in[1];   // [1024,1024]
  const float* Wk   = (const float*)d_in[2];
  const float* Wv   = (const float*)d_in[3];
  const float* Wo   = (const float*)d_in[4];   // [1024,1024] (D, HV)
  const float* emb0 = (const float*)d_in[5];
  const float* emb1 = (const float*)d_in[6];
  float* out = (float*)d_out;

  char* ws = (char*)d_ws;
  __bf16* Xb  = (__bf16*)ws;  ws += (size_t)ROWS * DMODEL * sizeof(__bf16);
  __bf16* Wqb = (__bf16*)ws;  ws += (size_t)DMODEL * DMODEL * sizeof(__bf16);
  __bf16* Wkb = (__bf16*)ws;  ws += (size_t)DMODEL * DMODEL * sizeof(__bf16);
  __bf16* Wvb = (__bf16*)ws;  ws += (size_t)DMODEL * DMODEL * sizeof(__bf16);
  __bf16* Wob = (__bf16*)ws;  ws += (size_t)DMODEL * DMODEL * sizeof(__bf16);
  float*  Qf  = (float*)ws;   ws += (size_t)ROWS * DMODEL * sizeof(float);
  float*  Kf  = (float*)ws;   ws += (size_t)ROWS * DMODEL * sizeof(float);
  float*  Vf  = (float*)ws;   ws += (size_t)ROWS * DMODEL * sizeof(float);
  __bf16* Gb  = (__bf16*)ws;  ws += (size_t)ROWS * DMODEL * sizeof(__bf16);

  const int nX = ROWS * DMODEL;       // 8,388,608
  const int nW = DMODEL * DMODEL;     // 1,048,576
  cvt_f32_bf16<<<nX / (4 * 256), 256, 0, stream>>>(X,  Xb,  nX);
  cvt_f32_bf16<<<nW / (4 * 256), 256, 0, stream>>>(Wq, Wqb, nW);
  cvt_f32_bf16<<<nW / (4 * 256), 256, 0, stream>>>(Wk, Wkb, nW);
  cvt_f32_bf16<<<nW / (4 * 256), 256, 0, stream>>>(Wv, Wvb, nW);
  cvt_f32_bf16<<<nW / (4 * 256), 256, 0, stream>>>(Wo, Wob, nW);

  // (ROWS/32)*(DMODEL/64) wave-tiles, 8 waves per 256-thread block
  const int gemmBlocks = (ROWS / 32) * (DMODEL / 64) / 8;   // 512
  gemm_wmma_bf16<<<gemmBlocks, 256, 0, stream>>>(Xb, Wqb, Qf, ROWS, DMODEL, DMODEL, 1);
  gemm_wmma_bf16<<<gemmBlocks, 256, 0, stream>>>(Xb, Wkb, Kf, ROWS, DMODEL, DMODEL, 1);
  gemm_wmma_bf16<<<gemmBlocks, 256, 0, stream>>>(Xb, Wvb, Vf, ROWS, DMODEL, DMODEL, 1);

  rosa_window_gate<<<(ROWS * NHEADS) / 256, 256, 0, stream>>>(Qf, Kf, Vf, emb0, emb1, Gb);

  gemm_wmma_bf16<<<gemmBlocks, 256, 0, stream>>>(Gb, Wob, out, ROWS, DMODEL, DMODEL, 0);
}
